// SeHGNNEncoder_38285338476562
// MI455X (gfx1250) — matrix operand order repaired and checked by reading, not compiled, not run on hardware
//
#include <hip/hip_runtime.h>
#include <hip/hip_bf16.h>

typedef __attribute__((ext_vector_type(16))) _Float16 v16h;
typedef __attribute__((ext_vector_type(8)))  float    v8f;

#define HID 256
#define NHOP 4

// ---------------------------------------------------------------- utilities

__global__ void fill_f32(float* p, long n) {
  long i = (long)blockIdx.x * blockDim.x + threadIdx.x;
  if (i < n) p[i] = 0.f;
}

// Pack W (KD x ND, f32 row-major) into WMMA B-fragment order, f16:
//   flat[(((nt*(KD/32) + k0)*32 + lane)*16 + e]  =  W[(k0*32 + (lane>>4)*16 + e)*ND + nt*16 + (lane&15)]
// so a wave's B fragment is one contiguous 32B load per lane.
__global__ void pack_w(const float* __restrict__ W, _Float16* __restrict__ out,
                       int KD, int ND) {
  int idx = blockIdx.x * blockDim.x + threadIdx.x;
  if (idx >= KD * ND) return;
  int e    = idx & 15;
  int lane = (idx >> 4) & 31;
  int t    = idx >> 9;              // (nt*(KD/32) + k0)
  int kt   = KD >> 5;               // K tiles
  int k0   = t % kt;
  int nt   = t / kt;
  int kk = k0 * 32 + (lane >> 4) * 16 + e;
  int n  = nt * 16 + (lane & 15);
  out[idx] = (_Float16)W[(long)kk * ND + n];
}

// ------------------------------------------------------- graph propagation

__global__ void degree_kernel(const int* __restrict__ row, const int* __restrict__ col,
                              float* ds, float* dd, int E) {
  int e = blockIdx.x * blockDim.x + threadIdx.x;
  if (e < E) {
    atomicAdd(&ds[row[e]], 1.f);
    atomicAdd(&dd[col[e]], 1.f);
  }
}

__global__ void weight_kernel(const int* __restrict__ row, const int* __restrict__ col,
                              const float* __restrict__ ds, const float* __restrict__ dd,
                              float* __restrict__ w, int E) {
  int e = blockIdx.x * blockDim.x + threadIdx.x;
  if (e < E) {
    float a = ds[row[e]], b = dd[col[e]];
    float ai = (a > 0.f) ? rsqrtf(a) : 0.f;
    float bi = (b > 0.f) ? rsqrtf(b) : 0.f;
    w[e] = ai * bi;
  }
}

// raw layout: (N, NHOP, HID). slice 0 = x, slices 1..3 zeroed for atomics.
__global__ void init_raw(const float* __restrict__ x, float* __restrict__ raw, int N) {
  int i = blockIdx.x * blockDim.x + threadIdx.x;
  if (i < N * HID) {
    int n = i / HID, d = i - n * HID;
    long base = (long)n * (NHOP * HID) + d;
    raw[base]           = x[i];
    raw[base + HID]     = 0.f;
    raw[base + 2 * HID] = 0.f;
    raw[base + 3 * HID] = 0.f;
  }
}

// dst[col][hop+1] += src[row][hop] * w  (8 threads per edge, 32 dims each)
__global__ void prop_kernel(const int* __restrict__ row, const int* __restrict__ col,
                            const float* __restrict__ w,
                            const float* __restrict__ src, float* __restrict__ dst,
                            int hop, int E) {
  int t = blockIdx.x * blockDim.x + threadIdx.x;
  int e = t >> 3;
  if (e >= E) return;
  int c = (t & 7) * 32;
  float wv = w[e];
  const float* s = src + (long)row[e] * (NHOP * HID) + hop * HID + c;
  float*       d = dst + (long)col[e] * (NHOP * HID) + (hop + 1) * HID + c;
#pragma unroll
  for (int i = 0; i < 32; ++i) atomicAdd(&d[i], s[i] * wv);
}

// ------------------------------------------------------- tiny 4x4 attention
// qkv: (4*N, 768) f16 rows [q|k|v]; out: (4*N, 256) f16.
// one thread per (node, head, query-pos)
__global__ void attn_kernel(const _Float16* __restrict__ qkv,
                            _Float16* __restrict__ out, int Nnodes) {
  int t = blockIdx.x * blockDim.x + threadIdx.x;
  if (t >= Nnodes * 16) return;
  int node = t >> 4;
  int h    = (t >> 2) & 3;
  int si   = t & 3;
  long rb = (long)node * 4;
  const _Float16* qrow = qkv + (rb + si) * 768 + h * 64;
  float q[64];
#pragma unroll
  for (int d = 0; d < 64; ++d) q[d] = (float)qrow[d];
  float sc[4];
  float mx = -1e30f;
#pragma unroll
  for (int j = 0; j < 4; ++j) {
    const _Float16* krow = qkv + (rb + j) * 768 + 256 + h * 64;
    float s = 0.f;
#pragma unroll
    for (int d = 0; d < 64; ++d) s += q[d] * (float)krow[d];
    s *= 0.125f;                   // 1/sqrt(64)
    sc[j] = s;
    mx = fmaxf(mx, s);
  }
  float sum = 0.f;
#pragma unroll
  for (int j = 0; j < 4; ++j) { sc[j] = __expf(sc[j] - mx); sum += sc[j]; }
  float inv = 1.f / sum;
  _Float16* orow = out + (rb + si) * 256 + h * 64;
  for (int d = 0; d < 64; ++d) {
    float o = 0.f;
#pragma unroll
    for (int j = 0; j < 4; ++j)
      o += sc[j] * (float)qkv[(rb + j) * 768 + 512 + h * 64 + d];
    orow[d] = (_Float16)(o * inv);
  }
}

// ----------------------------------------------------------- WMMA GEMM slab
// One workgroup (512 thr = 16 waves) computes a 16-row x ND-col slab of
//   C = A(16 x KD) @ W(KD x ND) + epilogue.
// Wave w owns column tiles (w + 16*r)*16, r < ND/256; K in steps of 32 with
// v_wmma_f32_16x16x32_f16. W is pre-packed (pack_w) so a B fragment is one
// 32-byte contiguous load per lane (2x global_load_b128).
//
// amode: 0 = A f16 row-major; 1 = A f32 row-major; 2 = A = mean over 4 hop
//        rows of f32 (N,4,KD) (out-head pooling).
// epi:   0 = +bias, store f16 (f32 if given)
//        4 = +bias +hop_emb[row%4], store f32 + f16
//        1 = +bias, exact GELU, store f16
//        2 = +bias +resid, LayerNorm(lng,lnb), store f32 (+f16)   [ND==256]
//        3 = +bias, row L2-normalize, store f32                   [ND==256]
template<int KD, int ND>
__launch_bounds__(512)
__global__ void gemm_slab(const void* __restrict__ Aptr, int amode,
                          const _Float16* __restrict__ W,   // packed
                          const float* __restrict__ bias,
                          const float* __restrict__ hop,
                          const float* __restrict__ resid,
                          const float* __restrict__ lng,
                          const float* __restrict__ lnb,
                          float* __restrict__ out32, _Float16* __restrict__ out16,
                          int M, int epi) {
  constexpr int NPT = ND / 256;                 // column tiles per wave
  constexpr int KT  = KD / 32;                  // K steps
  __shared__ _Float16 sA[16][KD + 8];
  __shared__ float    sR[16][HID + 8];          // slab staging for LN / L2

  const int tid  = threadIdx.x;
  const int wave = tid >> 5;
  const int lane = tid & 31;
  const int m0   = blockIdx.x * 16;

  // ---- cooperative A staging into LDS (clamped rows for the ragged tail)
  for (int idx = tid; idx < 16 * KD; idx += 512) {
    int r = idx / KD, k = idx - r * KD;
    int g = m0 + r; if (g >= M) g = M - 1;
    float v;
    if (amode == 0) {
      v = (float)((const _Float16*)Aptr)[(long)g * KD + k];
    } else if (amode == 1) {
      v = ((const float*)Aptr)[(long)g * KD + k];
    } else {
      const float* x = (const float*)Aptr + (long)g * 4 * KD + k;
      v = 0.25f * (x[0] + x[KD] + x[2 * KD] + x[3 * KD]);
    }
    sA[r][k] = (_Float16)v;
  }
  __syncthreads();

  const int mlo = lane & 15;                    // matrix row (A) / col (B/C)
  const int kh  = lane >> 4;                    // K-half select

  const v16h* __restrict__ Wp = (const v16h*)W; // packed fragments, 32B/lane

  v8f acc[NPT];
#pragma unroll
  for (int r = 0; r < NPT; ++r)
#pragma unroll
    for (int i = 0; i < 8; ++i) acc[r][i] = 0.f;

  for (int k0 = 0; k0 < KT; ++k0) {
    // A fragment: 1 contiguous 32B run in LDS (k = 16*kh + s) -> 2x ds_load_b128
    v16h a;
    const int ab = k0 * 32 + kh * 16;
#pragma unroll
    for (int i = 0; i < 8; ++i) {
      a[i]     = sA[mlo][ab + i];
      a[8 + i] = sA[mlo][ab + 8 + i];
    }
    if (k0 + 1 < KT)   // cover L2 latency for next K step's B fragment
      __builtin_prefetch((const void*)&Wp[((long)wave * KT + (k0 + 1)) * 32 + lane], 0, 1);
#pragma unroll
    for (int r = 0; r < NPT; ++r) {
      v16h b = Wp[((long)(wave + 16 * r) * KT + k0) * 32 + lane];
      acc[r] = __builtin_amdgcn_wmma_f32_16x16x32_f16(
          false, a, false, b, (short)0, acc[r], false, false);
    }
  }

  // ---- epilogues. C layout: acc[i] is row m = i + 8*kh, col = tile*16 + mlo
  if (epi == 2 || epi == 3) {
    // ND == 256: stage slab (+bias) to LDS, then per-row normalize
    const int n = wave * 16 + mlo;
#pragma unroll
    for (int i = 0; i < 8; ++i)
      sR[i + 8 * kh][n] = acc[0][i] + bias[n];
    __syncthreads();
    // wave w owns row w; residual loads are lane-contiguous (coalesced)
    const int row  = wave;
    const int g    = m0 + row;
    const bool live = (g < M);
    const long gbase = (long)(live ? g : 0) * 256;
    float vals[8];
    float s1 = 0.f, s2 = 0.f;
#pragma unroll
    for (int j = 0; j < 8; ++j) {
      int idx = lane + 32 * j;
      float v = sR[row][idx];
      if (epi == 2) v += resid[gbase + idx];
      vals[j] = v;
      s1 += v; s2 += v * v;
    }
#pragma unroll
    for (int off = 16; off > 0; off >>= 1) {
      s1 += __shfl_xor(s1, off, 32);
      s2 += __shfl_xor(s2, off, 32);
    }
    if (epi == 2) {
      float mean = s1 * (1.f / 256.f);
      float var  = s2 * (1.f / 256.f) - mean * mean;
      float rs   = rsqrtf(var + 1e-5f);
      if (live) {
#pragma unroll
        for (int j = 0; j < 8; ++j) {
          int idx = lane + 32 * j;
          float v = (vals[j] - mean) * rs * lng[idx] + lnb[idx];
          out32[gbase + idx] = v;
          if (out16) out16[gbase + idx] = (_Float16)v;
        }
      }
    } else {  // epi == 3 : L2 normalize into final output
      float scl = 1.f / fmaxf(sqrtf(s2), 1e-12f);
      if (live) {
#pragma unroll
        for (int j = 0; j < 8; ++j) {
          int idx = lane + 32 * j;
          out32[gbase + idx] = vals[j] * scl;
        }
      }
    }
  } else {
    // epi 0 / 1 / 4: elementwise epilogue, uniform full/tail split
    float vals[NPT][8];
#pragma unroll
    for (int r = 0; r < NPT; ++r) {
      const int n = (wave + 16 * r) * 16 + mlo;
#pragma unroll
      for (int i = 0; i < 8; ++i) {
        float v = acc[r][i] + bias[n];
        if (epi == 4) v += hop[((i + 8 * kh) & 3) * ND + n]; // hop = row % 4
        if (epi == 1) v = 0.5f * v * (1.f + erff(v * 0.70710678f));
        vals[r][i] = v;
      }
    }
    if (m0 + 16 <= M) {
#pragma unroll
      for (int r = 0; r < NPT; ++r) {
        const int n = (wave + 16 * r) * 16 + mlo;
#pragma unroll
        for (int i = 0; i < 8; ++i) {
          const long g = m0 + i + 8 * kh;
          if (out32) out32[g * ND + n] = vals[r][i];
          if (out16) out16[g * ND + n] = (_Float16)vals[r][i];
        }
      }
    } else {
#pragma unroll
      for (int r = 0; r < NPT; ++r) {
        const int n = (wave + 16 * r) * 16 + mlo;
#pragma unroll
        for (int i = 0; i < 8; ++i) {
          const long g = m0 + i + 8 * kh;
          if (g >= M) continue;
          if (out32) out32[g * ND + n] = vals[r][i];
          if (out16) out16[g * ND + n] = (_Float16)vals[r][i];
        }
      }
    }
  }
}

// ------------------------------------------------------------------- driver

extern "C" void kernel_launch(void* const* d_in, const int* in_sizes, int n_in,
                              void* d_out, int out_size, void* d_ws, size_t ws_size,
                              hipStream_t stream) {
  (void)n_in; (void)out_size; (void)ws_size;

  const float* x_p  = (const float*)d_in[0];
  const float* x_c  = (const float*)d_in[1];
  const int*   e_pc = (const int*)d_in[2];
  const int*   e_cp = (const int*)d_in[3];
  const int NP = in_sizes[0] / HID;
  const int NC = in_sizes[1] / HID;
  const int E1 = in_sizes[2] / 2;
  const int E2 = in_sizes[3] / 2;

  int pi = 4;
  const float* in_W_p  = (const float*)d_in[pi++];
  const float* in_b_p  = (const float*)d_in[pi++];
  const float* in_W_c  = (const float*)d_in[pi++];
  const float* in_b_c  = (const float*)d_in[pi++];
  const float* hop_emb = (const float*)d_in[pi++];
  const float* Wqkv    = (const float*)d_in[pi++];
  const float* bqkv    = (const float*)d_in[pi++];
  const float* Wo      = (const float*)d_in[pi++];
  const float* bo      = (const float*)d_in[pi++];
  const float* ln1g    = (const float*)d_in[pi++];
  const float* ln1b    = (const float*)d_in[pi++];
  const float* W1      = (const float*)d_in[pi++];
  const float* b1      = (const float*)d_in[pi++];
  const float* W2      = (const float*)d_in[pi++];
  const float* b2      = (const float*)d_in[pi++];
  const float* ln2g    = (const float*)d_in[pi++];
  const float* ln2b    = (const float*)d_in[pi++];
  const float* out_W_p = (const float*)d_in[pi++];
  const float* out_b_p = (const float*)d_in[pi++];
  const float* out_W_c = (const float*)d_in[pi++];
  const float* out_b_c = (const float*)d_in[pi++];

  // ---- workspace partition (bump allocator, 256B aligned)
  char* ws = (char*)d_ws;
  size_t off = 0;
  auto alloc = [&](size_t bytes) -> char* {
    char* p = ws + off;
    off = (off + bytes + 255) & ~(size_t)255;
    return p;
  };

  float* degs = (float*)alloc((size_t)2 * (NP + NC) * 4);
  float* dsp = degs;                 // project out-degree (edge_pc rows)
  float* ddc = degs + NP;            // company in-degree  (edge_pc cols)
  float* dsc = degs + NP + NC;       // company out-degree (edge_cp rows)
  float* ddp = degs + NP + 2 * NC;   // project in-degree  (edge_cp cols)
  float* w_pc = (float*)alloc((size_t)E1 * 4);
  float* w_cp = (float*)alloc((size_t)E2 * 4);

  _Float16* winp16  = (_Float16*)alloc(65536 * 2);
  _Float16* winc16  = (_Float16*)alloc(65536 * 2);
  _Float16* wqkv16  = (_Float16*)alloc(196608 * 2);
  _Float16* wo16    = (_Float16*)alloc(65536 * 2);
  _Float16* w1_16   = (_Float16*)alloc(131072 * 2);
  _Float16* w2_16   = (_Float16*)alloc(131072 * 2);
  _Float16* woutp16 = (_Float16*)alloc(65536 * 2);
  _Float16* woutc16 = (_Float16*)alloc(65536 * 2);

  float* raw_p = (float*)alloc((size_t)NP * NHOP * HID * 4);
  float* raw_c = (float*)alloc((size_t)NC * NHOP * HID * 4);

  const long MR = 4L * (NP > NC ? NP : NC);     // max GEMM rows
  // h0 region (reused for FFN activations once h0 is dead)
  char* h0base = alloc((size_t)MR * 1536);
  float*    h0f32 = (float*)h0base;             // MR*1024 B
  _Float16* h0f16 = (_Float16*)(h0base + MR * 1024);
  _Float16* ffn16 = (_Float16*)h0base;          // MR*1024 B, alias (h0 dead)
  // qkv region (reused for x1 once qkv is dead)
  char* qbase = alloc((size_t)MR * 1536);
  _Float16* qkv16 = (_Float16*)qbase;           // MR*1536 B
  float*    x1f32 = (float*)qbase;              // MR*1024 B, alias
  _Float16* x1f16 = (_Float16*)(qbase + MR * 1024);
  _Float16* attn16 = (_Float16*)alloc((size_t)MR * 512);
  float*    x2f32  = (float*)alloc((size_t)MR * 1024);

  // ---- graph normalization weights
  {
    long nd = 2L * (NP + NC);
    fill_f32<<<(int)((nd + 255) / 256), 256, 0, stream>>>(degs, nd);
  }
  degree_kernel<<<(E1 + 255) / 256, 256, 0, stream>>>(e_pc, e_pc + E1, dsp, ddc, E1);
  degree_kernel<<<(E2 + 255) / 256, 256, 0, stream>>>(e_cp, e_cp + E2, dsc, ddp, E2);
  weight_kernel<<<(E1 + 255) / 256, 256, 0, stream>>>(e_pc, e_pc + E1, dsp, ddc, w_pc, E1);
  weight_kernel<<<(E2 + 255) / 256, 256, 0, stream>>>(e_cp, e_cp + E2, dsc, ddp, w_cp, E2);

  // ---- K-hop propagation into raw (N, 4, 256)
  init_raw<<<(NP * HID + 255) / 256, 256, 0, stream>>>(x_p, raw_p, NP);
  init_raw<<<(NC * HID + 255) / 256, 256, 0, stream>>>(x_c, raw_c, NC);
  for (int h = 0; h < 3; ++h) {
    prop_kernel<<<(E1 * 8 + 255) / 256, 256, 0, stream>>>(e_pc, e_pc + E1, w_pc,
                                                          raw_p, raw_c, h, E1);
    prop_kernel<<<(E2 * 8 + 255) / 256, 256, 0, stream>>>(e_cp, e_cp + E2, w_cp,
                                                          raw_c, raw_p, h, E2);
  }

  // ---- weight pack: f32 -> f16 in WMMA B-fragment order (L2-resident)
  pack_w<<<(65536 + 255) / 256, 256, 0, stream>>>(in_W_p, winp16, 256, 256);
  pack_w<<<(65536 + 255) / 256, 256, 0, stream>>>(in_W_c, winc16, 256, 256);
  pack_w<<<(196608 + 255) / 256, 256, 0, stream>>>(Wqkv, wqkv16, 256, 768);
  pack_w<<<(65536 + 255) / 256, 256, 0, stream>>>(Wo, wo16, 256, 256);
  pack_w<<<(131072 + 255) / 256, 256, 0, stream>>>(W1, w1_16, 256, 512);
  pack_w<<<(131072 + 255) / 256, 256, 0, stream>>>(W2, w2_16, 512, 256);
  pack_w<<<(65536 + 255) / 256, 256, 0, stream>>>(out_W_p, woutp16, 256, 256);
  pack_w<<<(65536 + 255) / 256, 256, 0, stream>>>(out_W_c, woutc16, 256, 256);

  // ---- per-type transformer head
  auto run_head = [&](const float* raw, const _Float16* winT, const float* binT,
                      const _Float16* woutT, const float* boutT,
                      int Nt, float* outPtr) {
    const int Mr = 4 * Nt;
    const int gm = (Mr + 15) / 16;
    // h0 = raw @ inW + in_b + hop_emb
    gemm_slab<256, 256><<<gm, 512, 0, stream>>>(raw, 1, winT, binT, hop_emb,
        nullptr, nullptr, nullptr, h0f32, h0f16, Mr, 4);
    // qkv = h0 @ Wqkv + bqkv
    gemm_slab<256, 768><<<gm, 512, 0, stream>>>(h0f16, 0, wqkv16, bqkv, nullptr,
        nullptr, nullptr, nullptr, nullptr, qkv16, Mr, 0);
    // 4x4 softmax attention, 4 heads of 64
    attn_kernel<<<(Nt * 16 + 255) / 256, 256, 0, stream>>>(qkv16, attn16, Nt);
    // x1 = LN(h0 + attn @ Wo + bo)
    gemm_slab<256, 256><<<gm, 512, 0, stream>>>(attn16, 0, wo16, bo, nullptr,
        h0f32, ln1g, ln1b, x1f32, x1f16, Mr, 2);
    // f = gelu(x1 @ W1 + b1)
    gemm_slab<256, 512><<<gm, 512, 0, stream>>>(x1f16, 0, w1_16, b1, nullptr,
        nullptr, nullptr, nullptr, nullptr, ffn16, Mr, 1);
    // x2 = LN(x1 + f @ W2 + b2)
    gemm_slab<512, 256><<<gm, 512, 0, stream>>>(ffn16, 0, w2_16, b2, nullptr,
        x1f32, ln2g, ln2b, x2f32, nullptr, Mr, 2);
    // out = l2norm(mean_s(x2) @ outW + out_b)
    const int gn = (Nt + 15) / 16;
    gemm_slab<256, 256><<<gn, 512, 0, stream>>>(x2f32, 2, woutT, boutT, nullptr,
        nullptr, nullptr, nullptr, outPtr, nullptr, Nt, 3);
  };

  run_head(raw_p, winp16, in_b_p, woutp16, out_b_p, NP, (float*)d_out);
  run_head(raw_c, winc16, in_b_c, woutc16, out_b_c, NC,
           (float*)d_out + (size_t)NP * HID);
}
